// Attention_46840913330813
// MI455X (gfx1250) — matrix operand order repaired and checked by reading, not compiled, not run on hardware
//
#include <hip/hip_runtime.h>
#include <hip/hip_bf16.h>

// ---------------------------------------------------------------------------
// MHA layer for MI455X (gfx1250, wave32, WMMA + TDM).
// All matmuls via v_wmma_f32_16x16x32_bf16 after one fp32->bf16 pass.
// GEMM: workgroup shares a 32x128 A panel staged by tensor_load_to_lds (TDM),
// each wave computes a 32x64 output tile (8 WMMA / 32-K step).
// ---------------------------------------------------------------------------

typedef __attribute__((ext_vector_type(8)))  __bf16       v8bf;
typedef __attribute__((ext_vector_type(16))) __bf16       v16bf;
typedef __attribute__((ext_vector_type(8)))  float        v8f;
typedef __attribute__((ext_vector_type(4)))  unsigned int u32x4;
typedef __attribute__((ext_vector_type(8)))  unsigned int u32x8;

#define HID   2560
#define NHEAD 32
#define HDIM  80
#define SEQ   2048
#define BATCH 2
#define MTOK  (BATCH * SEQ)   // 4096
#define ROTD  20

static __device__ __forceinline__ v8bf ld8(const __bf16* p) {
  return *reinterpret_cast<const v8bf*>(p);
}
static __device__ __forceinline__ v8bf zero8() {
  v8bf z;
#pragma unroll
  for (int i = 0; i < 8; ++i) z[i] = (__bf16)0.0f;
  return z;
}
static __device__ __forceinline__ v8f zero8f() {
  v8f z;
#pragma unroll
  for (int i = 0; i < 8; ++i) z[i] = 0.0f;
  return z;
}
static __device__ __forceinline__ v16bf cat16(v8bf lo, v8bf hi) {
  return __builtin_shufflevector(lo, hi, 0,1,2,3,4,5,6,7,8,9,10,11,12,13,14,15);
}
static __device__ __forceinline__ v8f wmma_bf16(v16bf a, v16bf b, v8f c) {
  return __builtin_amdgcn_wmma_f32_16x16x32_bf16(false, a, false, b,
                                                 (short)0, c, false, false);
}
static __device__ __forceinline__ unsigned int rfl(unsigned int x) {
  return (unsigned int)__builtin_amdgcn_readfirstlane((int)x);
}

// ---------------------------------------------------------------------------
// fp32 -> bf16 straight convert
// ---------------------------------------------------------------------------
__global__ void convert_kernel(const float* __restrict__ src,
                               __bf16* __restrict__ dst, int n) {
  int i = blockIdx.x * blockDim.x + threadIdx.x;
  if (i < n) dst[i] = (__bf16)src[i];
}

// ---------------------------------------------------------------------------
// W[k][n] fp32  ->  Wt[n][k] bf16   (32x32 LDS tile transpose; dims divide)
// ---------------------------------------------------------------------------
__global__ void wtrans_kernel(const float* __restrict__ Wf,
                              __bf16* __restrict__ Wt, int K, int N) {
  __shared__ float tile[32][33];
  int k = blockIdx.y * 32 + threadIdx.y;
  int n = blockIdx.x * 32 + threadIdx.x;
  tile[threadIdx.y][threadIdx.x] = Wf[(size_t)k * N + n];
  __syncthreads();
  int nn = blockIdx.x * 32 + threadIdx.y;
  int kk = blockIdx.y * 32 + threadIdx.x;
  Wt[(size_t)nn * K + kk] = (__bf16)tile[threadIdx.x][threadIdx.y];
}

// ---------------------------------------------------------------------------
// GEMM: C[M][N] = A[M][K] * Bt[N][K]^T, bf16 in, f32 accumulate.
// Grid (M/32, N/512); block 256 = 8 waves. Workgroup tile 32x512:
//   - A panel 32 rows x 128 K staged into LDS by the Tensor Data Mover
//   - each wave: 32x64 strip = 2 row-tiles x 4 col-tiles, 8 WMMA per 32-K step
// ---------------------------------------------------------------------------
template <bool OUT_BF16>
__global__ __launch_bounds__(256) void gemm_wmma(const __bf16* __restrict__ A,
                                                 const __bf16* __restrict__ Bt,
                                                 void* __restrict__ Cout,
                                                 int M, int N, int K) {
  __shared__ __align__(16) __bf16 Atile[32 * 128];   // 8 KB panel

  const int lane = threadIdx.x & 31;
  const int wave = threadIdx.x >> 5;
  const int m0 = blockIdx.x * 32;
  const int n0 = blockIdx.y * 512 + wave * 64;
  const int ac0 = (lane < 16) ? 0 : 16;  // A-frag K base per lane half
  const int bc0 = (lane < 16) ? 0 : 16;  // B-frag K base per lane half

  const __bf16* bp[4];
#pragma unroll
  for (int t = 0; t < 4; ++t)
    bp[t] = Bt + (size_t)(n0 + t * 16 + (lane & 15)) * K;

  v8f c[2][4];
#pragma unroll
  for (int rt = 0; rt < 2; ++rt)
#pragma unroll
    for (int t = 0; t < 4; ++t) c[rt][t] = zero8f();

  // ---- constant parts of the TDM descriptor (D#) ----
  const unsigned int ldsOff = (unsigned int)(size_t)(__bf16*)Atile;
  // group1: data_size=2B; tensor_dim0=K, tensor_dim1=M; tile 128x32; stride0=K
  const unsigned int g1w0 = 1u << 16;                       // data_size=1 (2B)
  const unsigned int g1w1 = ((unsigned int)K & 0xFFFFu) << 16;
  const unsigned int g1w2 = (((unsigned int)K >> 16) & 0xFFFFu) |
                            (((unsigned int)M & 0xFFFFu) << 16);
  const unsigned int g1w3 = (((unsigned int)M >> 16) & 0xFFFFu) | (128u << 16);
  const unsigned int g1w4 = 32u;                            // tile_dim1=32 rows
  const unsigned int g1w5 = (unsigned int)K;                // dim0 stride lo
  const unsigned int g1w6 = 0u;
  const unsigned int g1w7 = 0u;

  for (int kc = 0; kc < K; kc += 128) {
    __syncthreads();   // previous panel fully consumed before TDM overwrites
    if (wave == 0) {
      unsigned long long ga =
          (unsigned long long)(const void*)(A + (size_t)m0 * K + kc);
      u32x4 g0;
      g0[0] = rfl(1u);                                   // count=1, load, user
      g0[1] = rfl(ldsOff);                               // lds_addr
      g0[2] = rfl((unsigned int)ga);                     // global_addr[31:0]
      g0[3] = rfl(((unsigned int)(ga >> 32) & 0x01FFFFFFu) | (2u << 30)); // +type=2
      u32x8 g1;
      g1[0] = rfl(g1w0); g1[1] = rfl(g1w1); g1[2] = rfl(g1w2); g1[3] = rfl(g1w3);
      g1[4] = rfl(g1w4); g1[5] = rfl(g1w5); g1[6] = rfl(g1w6); g1[7] = rfl(g1w7);
      asm volatile("tensor_load_to_lds %0, %1" :: "s"(g0), "s"(g1) : "memory");
      __builtin_amdgcn_s_wait_tensorcnt(0);
    }
    __syncthreads();   // panel visible to all waves

#pragma unroll
    for (int kk32 = 0; kk32 < 4; ++kk32) {
      const int kb = kk32 * 32;
      v16bf a[2];
#pragma unroll
      for (int rt = 0; rt < 2; ++rt) {
        const __bf16* as = &Atile[(rt * 16 + (lane & 15)) * 128 + kb + ac0];
        a[rt] = cat16(ld8(as), ld8(as + 8));             // ds_load_b128 x2
      }
#pragma unroll
      for (int t = 0; t < 4; ++t) {
        const __bf16* bq = bp[t] + kc + kb + bc0;
        v16bf b = cat16(ld8(bq), ld8(bq + 8));
        c[0][t] = wmma_bf16(a[0], b, c[0][t]);
        c[1][t] = wmma_bf16(a[1], b, c[1][t]);
      }
    }
    if (kc + 256 <= K) __builtin_prefetch(bp[0] + kc + 128, 0, 0); // global_prefetch
  }

#pragma unroll
  for (int rt = 0; rt < 2; ++rt)
#pragma unroll
    for (int t = 0; t < 4; ++t)
#pragma unroll
      for (int r = 0; r < 8; ++r) {
        int row = m0 + rt * 16 + r + ((lane < 16) ? 0 : 8);
        int col = n0 + t * 16 + (lane & 15);
        if (OUT_BF16)
          ((__bf16*)Cout)[(size_t)row * N + col] = (__bf16)c[rt][t][r];
        else
          ((float*)Cout)[(size_t)row * N + col] = c[rt][t][r];
      }
}

// ---------------------------------------------------------------------------
// RoPE in place on first 20 dims of each head of Q and K (pair (i, i+10)).
// ---------------------------------------------------------------------------
__global__ void rope_kernel(__bf16* __restrict__ q, __bf16* __restrict__ k,
                            const int* __restrict__ pos_ids) {
  int idx = blockIdx.x * blockDim.x + threadIdx.x;
  const int total = BATCH * SEQ * NHEAD * (ROTD / 2);
  if (idx >= total) return;
  int i = idx % (ROTD / 2);
  int t = idx / (ROTD / 2);
  int hh = t % NHEAD; t /= NHEAD;
  int s = t % SEQ;
  int b = t / SEQ;
  float pos = (float)pos_ids[b * SEQ + s];
  float freq = __expf(-((float)(2 * i) / (float)ROTD) * 9.210340372f);
  float ang = pos * freq;
  float cs = __cosf(ang), sn = __sinf(ang);
  size_t base = (size_t)(b * SEQ + s) * HID + hh * HDIM;
  float q1 = (float)q[base + i], q2 = (float)q[base + i + ROTD / 2];
  q[base + i]            = (__bf16)(q1 * cs - q2 * sn);
  q[base + i + ROTD / 2] = (__bf16)(q2 * cs + q1 * sn);
  float k1 = (float)k[base + i], k2 = (float)k[base + i + ROTD / 2];
  k[base + i]            = (__bf16)(k1 * cs - k2 * sn);
  k[base + i + ROTD / 2] = (__bf16)(k2 * cs + k1 * sn);
}

// ---------------------------------------------------------------------------
// V[token][h*80+d] -> Vt[b][h][d][s]  (so PV B-fragments load contiguously)
// ---------------------------------------------------------------------------
__global__ void vtrans_kernel(const __bf16* __restrict__ v,
                              __bf16* __restrict__ vt) {
  int idx = blockIdx.x * blockDim.x + threadIdx.x;  // ((b*NH+h)*D+d)*S + s
  const int total = BATCH * NHEAD * HDIM * SEQ;
  if (idx >= total) return;
  int s = idx % SEQ;
  int t = idx / SEQ;
  int d = t % HDIM; t /= HDIM;
  int h = t % NHEAD;
  int b = t / NHEAD;
  vt[idx] = v[(size_t)(b * SEQ + s) * HID + h * HDIM + d];
}

// ---------------------------------------------------------------------------
// Flash attention. One wave per 16-row Q tile, 32 keys per j-step.
// 6 score WMMA + online softmax (shfl_xor -> ds_bpermute) + LDS re-layout of
// probs (C-frag -> A-frag) + 5 PV WMMA. No block barriers in divergent loop.
// ---------------------------------------------------------------------------
__global__ __launch_bounds__(256) void attn_kernel(const __bf16* __restrict__ qb,
                                                   const __bf16* __restrict__ kb,
                                                   const __bf16* __restrict__ vt,
                                                   __bf16* __restrict__ ob) {
  __shared__ __align__(16) __bf16 plds[8][16 * 32];

  const int lane = threadIdx.x & 31;
  const int wave = threadIdx.x >> 5;
  const int tile = blockIdx.x * 8 + wave;         // 8192 tiles total
  const int qt = tile & 127;                      // S/16 = 128
  const int bh = tile >> 7;
  const int h = bh & (NHEAD - 1);
  const int b = bh >> 5;
  const int q0 = qt * 16;

  const int ac0 = (lane < 16) ? 0 : 16;
  const int bc0 = (lane < 16) ? 0 : 16;
  const int ncol = lane & 15;

  const __bf16* qp = qb + (size_t)(b * SEQ + q0 + (lane & 15)) * HID + h * HDIM;
  v16bf qf[3];
#pragma unroll
  for (int c = 0; c < 3; ++c) {
    int kd = c * 32;
    v8bf lo = (kd + ac0 + 8 <= HDIM) ? ld8(qp + kd + ac0) : zero8();
    v8bf hi = (kd + ac0 + 16 <= HDIM) ? ld8(qp + kd + ac0 + 8) : zero8();
    qf[c] = cat16(lo, hi);
  }

  float m[8], l[8];
  v8f acc[5];
#pragma unroll
  for (int r = 0; r < 8; ++r) { m[r] = -3.0e38f; l[r] = 0.0f; }
#pragma unroll
  for (int t = 0; t < 5; ++t) acc[t] = zero8f();

  const float sscale = 0.11180339887f;  // 1/sqrt(80)
  const int qend = q0 + 15;

  for (int j0 = 0; j0 <= qend; j0 += 32) {
    v8f s0 = zero8f(), s1 = zero8f();
    const __bf16* kp0 = kb + (size_t)(b * SEQ + j0 + (lane & 15)) * HID + h * HDIM;
    const __bf16* kp1 = kp0 + (size_t)16 * HID;
#pragma unroll
    for (int c = 0; c < 3; ++c) {
      int off = c * 32 + bc0;
      v8bf b0lo = (off + 8  <= HDIM) ? ld8(kp0 + off)     : zero8();
      v8bf b0hi = (off + 16 <= HDIM) ? ld8(kp0 + off + 8) : zero8();
      v8bf b1lo = (off + 8  <= HDIM) ? ld8(kp1 + off)     : zero8();
      v8bf b1hi = (off + 16 <= HDIM) ? ld8(kp1 + off + 8) : zero8();
      s0 = wmma_bf16(qf[c], cat16(b0lo, b0hi), s0);
      s1 = wmma_bf16(qf[c], cat16(b1lo, b1hi), s1);
    }

#pragma unroll
    for (int r = 0; r < 8; ++r) {
      int qrow = q0 + r + ((lane < 16) ? 0 : 8);
      float a0 = (j0 + ncol      <= qrow) ? s0[r] * sscale : -3.0e38f;
      float a1 = (j0 + 16 + ncol <= qrow) ? s1[r] * sscale : -3.0e38f;
      float mt = fmaxf(a0, a1);
      mt = fmaxf(mt, __shfl_xor(mt, 1, 32));
      mt = fmaxf(mt, __shfl_xor(mt, 2, 32));
      mt = fmaxf(mt, __shfl_xor(mt, 4, 32));
      mt = fmaxf(mt, __shfl_xor(mt, 8, 32));
      float mnew = fmaxf(m[r], mt);
      float p0 = __expf(a0 - mnew);
      float p1 = __expf(a1 - mnew);
      float rs = p0 + p1;
      rs += __shfl_xor(rs, 1, 32);
      rs += __shfl_xor(rs, 2, 32);
      rs += __shfl_xor(rs, 4, 32);
      rs += __shfl_xor(rs, 8, 32);
      float corr = __expf(m[r] - mnew);
      l[r] = l[r] * corr + rs;
      m[r] = mnew;
#pragma unroll
      for (int t = 0; t < 5; ++t) acc[t][r] *= corr;
      int Mr = r + ((lane < 16) ? 0 : 8);
      plds[wave][Mr * 32 + ncol]      = (__bf16)p0;
      plds[wave][Mr * 32 + 16 + ncol] = (__bf16)p1;
    }

    asm volatile("s_wait_dscnt 0" ::: "memory");

    const __bf16* pl = &plds[wave][(lane & 15) * 32];
    v16bf pf = cat16(ld8(pl + ac0), ld8(pl + ac0 + 8));

#pragma unroll
    for (int t = 0; t < 5; ++t) {
      const __bf16* vp = vt + (size_t)((b * NHEAD + h) * HDIM + t * 16 + (lane & 15)) * SEQ
                         + j0 + bc0;
      v16bf vf = cat16(ld8(vp), ld8(vp + 8));
      acc[t] = wmma_bf16(pf, vf, acc[t]);
    }
  }

#pragma unroll
  for (int t = 0; t < 5; ++t) {
#pragma unroll
    for (int r = 0; r < 8; ++r) {
      int row = q0 + r + ((lane < 16) ? 0 : 8);
      float val = acc[t][r] / l[r];
      ob[(size_t)(b * SEQ + row) * HID + h * HDIM + t * 16 + ncol] = (__bf16)val;
    }
  }
}

// ---------------------------------------------------------------------------
// Host-side launch
// ---------------------------------------------------------------------------
static size_t align256(size_t x) { return (x + 255) & ~(size_t)255; }

extern "C" void kernel_launch(void* const* d_in, const int* in_sizes, int n_in,
                              void* d_out, int out_size, void* d_ws, size_t ws_size,
                              hipStream_t stream) {
  (void)in_sizes; (void)n_in; (void)out_size; (void)ws_size;
  const float* X  = (const float*)d_in[0];
  // d_in[1] = attention_mask (all ones) -> causal mask handled in-kernel
  const int*   pos = (const int*)d_in[2];
  const float* Wq = (const float*)d_in[3];
  const float* Wk = (const float*)d_in[4];
  const float* Wv = (const float*)d_in[5];
  const float* Wo = (const float*)d_in[6];
  float* out = (float*)d_out;

  const size_t sAct = (size_t)MTOK * HID * sizeof(__bf16);   // 21 MB
  const size_t sW   = (size_t)HID * HID * sizeof(__bf16);    // 13 MB

  char* w = (char*)d_ws;
  __bf16* Xb  = (__bf16*)w; w += align256(sAct);
  __bf16* Wqt = (__bf16*)w; w += align256(sW);
  __bf16* Wkt = (__bf16*)w; w += align256(sW);
  __bf16* Wvt = (__bf16*)w; w += align256(sW);
  __bf16* Wot = (__bf16*)w; w += align256(sW);
  __bf16* Qb  = (__bf16*)w; w += align256(sAct);
  __bf16* Kb  = (__bf16*)w; w += align256(sAct);
  __bf16* Vb  = (__bf16*)w; w += align256(sAct);
  __bf16* Vt  = (__bf16*)w; w += align256(sAct);
  __bf16* Ab  = Xb;  // reuse X bf16 buffer for attention output (X dead by then)

  const int nAct = MTOK * HID;  // 10,485,760

  convert_kernel<<<(nAct + 255) / 256, 256, 0, stream>>>(X, Xb, nAct);
  dim3 wtg(HID / 32, HID / 32), wtb(32, 32);
  wtrans_kernel<<<wtg, wtb, 0, stream>>>(Wq, Wqt, HID, HID);
  wtrans_kernel<<<wtg, wtb, 0, stream>>>(Wk, Wkt, HID, HID);
  wtrans_kernel<<<wtg, wtb, 0, stream>>>(Wv, Wvt, HID, HID);
  wtrans_kernel<<<wtg, wtb, 0, stream>>>(Wo, Wot, HID, HID);

  dim3 gg(MTOK / 32, HID / 512);
  gemm_wmma<true><<<gg, 256, 0, stream>>>(Xb, Wqt, Qb, MTOK, HID, HID);
  gemm_wmma<true><<<gg, 256, 0, stream>>>(Xb, Wkt, Kb, MTOK, HID, HID);
  gemm_wmma<true><<<gg, 256, 0, stream>>>(Xb, Wvt, Vb, MTOK, HID, HID);

  const int nRope = BATCH * SEQ * NHEAD * (ROTD / 2);
  rope_kernel<<<(nRope + 255) / 256, 256, 0, stream>>>(Qb, Kb, pos);

  vtrans_kernel<<<(nAct + 255) / 256, 256, 0, stream>>>(Vb, Vt);

  attn_kernel<<<(BATCH * NHEAD * (SEQ / 16)) / 8, 256, 0, stream>>>(Qb, Kb, Vt, Ab);

  gemm_wmma<false><<<gg, 256, 0, stream>>>(Ab, Wot, out, MTOK, HID, HID);
}
